// Attention_63058709840460
// MI455X (gfx1250) — matrix operand order repaired and checked by reading, not compiled, not run on hardware
//
#include <hip/hip_runtime.h>
#include <hip/hip_bf16.h>

typedef __bf16 bf16_t;
typedef __bf16 bf16x16 __attribute__((ext_vector_type(16)));
typedef __bf16 bf16x8  __attribute__((ext_vector_type(8)));
typedef float  f32x8   __attribute__((ext_vector_type(8)));

#define B_     8
#define NSEQ   1024
#define DIMM   512
#define HEADS_ 8
#define DH     64
#define INNER_ 512
#define NPOS   1025   // 2*MAX_POS+1

__device__ __forceinline__ f32x8 f32x8_zero() {
  f32x8 z = {0.f, 0.f, 0.f, 0.f, 0.f, 0.f, 0.f, 0.f};
  return z;
}

// A fragment (16x32, MxK), 16-bit: lane l -> M=l&15, g=l>>4.
// halves 0..7: K = k0 + 8g + 0..7 ; halves 8..15: K = k0 + 16 + 8g + 0..7
__device__ __forceinline__ bf16x16 load_a_frag(const bf16_t* base, int ld, int m0,
                                               int k0, int lane) {
  const int m = m0 + (lane & 15);
  const int g = lane >> 4;
  const bf16_t* p = base + (size_t)m * ld + k0 + 8 * g;
  bf16x8 lo = *(const bf16x8*)(p);
  bf16x8 hi = *(const bf16x8*)(p + 16);
  bf16x16 a;
#pragma unroll
  for (int i = 0; i < 8; ++i) { a[i] = lo[i]; a[8 + i] = hi[i]; }
  return a;
}

// B fragment (32x16, KxN), 16-bit: lane l -> N=l&15, g=l>>4.
// halves 0..7: K = k0 + 8g + 0..7 ; halves 8..15: K = k0 + 16 + 8g + 0..7. Storage is N-major with K contiguous.
__device__ __forceinline__ bf16x16 load_b_frag(const bf16_t* base, int ld, int n0,
                                               int k0, int lane) {
  const int n = n0 + (lane & 15);
  const int g = lane >> 4;
  const bf16_t* p = base + (size_t)n * ld + k0 + 8 * g;
  bf16x8 lo = *(const bf16x8*)(p);
  bf16x8 hi = *(const bf16x8*)(p + 16);
  bf16x16 b;
#pragma unroll
  for (int i = 0; i < 8; ++i) { b[i] = lo[i]; b[8 + i] = hi[i]; }
  return b;
}

__device__ __forceinline__ f32x8 wmma_bf16(bf16x16 a, bf16x16 b, f32x8 c) {
  return __builtin_amdgcn_wmma_f32_16x16x32_bf16(false, a, false, b, (short)0, c,
                                                 false, false);
}

// ---------------- conversion kernels ----------------

__global__ void cvt_f32_bf16_kernel(const float* __restrict__ src,
                                    bf16_t* __restrict__ dst, int n) {
  int i = blockIdx.x * blockDim.x + threadIdx.x;
  if (i < n) dst[i] = (bf16_t)src[i];
}

// wt[c][k] = combined [Wq | Wkv] column c, row k  (transposed, bf16)
__global__ void cvt_wqkv_kernel(const float* __restrict__ Wq,
                                const float* __restrict__ Wkv,
                                bf16_t* __restrict__ wt) {
  int idx = blockIdx.x * blockDim.x + threadIdx.x;
  if (idx >= 1536 * 512) return;
  int c = idx >> 9, k = idx & 511;
  float v = (c < 512) ? Wq[(size_t)k * 512 + c] : Wkv[(size_t)k * 1024 + (c - 512)];
  wt[idx] = (bf16_t)v;
}

__global__ void cvt_wo_kernel(const float* __restrict__ Wo, bf16_t* __restrict__ wot) {
  int idx = blockIdx.x * blockDim.x + threadIdx.x;
  if (idx >= 512 * 512) return;
  int c = idx >> 9, k = idx & 511;
  wot[idx] = (bf16_t)Wo[(size_t)k * 512 + c];
}

// ---------------- fused QKV projection ----------------
// x_bf16 [8192][512] @ Wt[1536][512]^T -> q [bh][n][64] (scaled by 0.125),
// k [bh][n][64], v transposed [bh][64][n]
__global__ __launch_bounds__(256) void qkv_gemm_kernel(
    const bf16_t* __restrict__ xb, const bf16_t* __restrict__ wt,
    bf16_t* __restrict__ qw, bf16_t* __restrict__ kw, bf16_t* __restrict__ vtw) {
  const int lane = threadIdx.x & 31;
  const int wave = threadIdx.x >> 5;
  const int ln = lane & 15;
  const int g = lane >> 4;
  const int m0 = blockIdx.x * 128 + wave * 16;

  bf16x16 A[16];
#pragma unroll
  for (int kc = 0; kc < 16; ++kc) A[kc] = load_a_frag(xb, DIMM, m0, 32 * kc, lane);

  for (int ct = 0; ct < 96; ++ct) {
    f32x8 c = f32x8_zero();
#pragma unroll
    for (int kc = 0; kc < 16; ++kc) {
      bf16x16 bf = load_b_frag(wt, DIMM, ct * 16, 32 * kc, lane);
      c = wmma_bf16(A[kc], bf, c);
    }
    const int col = ct * 16 + ln;
#pragma unroll
    for (int r = 0; r < 8; ++r) {
      const int row = m0 + r + 8 * g;
      const int b = row >> 10;
      const int n = row & 1023;
      const float v = c[r];
      if (col < INNER_) {
        const int h = col >> 6, d = col & 63;
        qw[(((size_t)b * HEADS_ + h) * NSEQ + n) * DH + d] = (bf16_t)(v * 0.125f);
      } else if (col < 2 * INNER_) {
        const int cc = col - INNER_;
        const int h = cc >> 6, d = cc & 63;
        kw[(((size_t)b * HEADS_ + h) * NSEQ + n) * DH + d] = (bf16_t)v;
      } else {
        const int cc = col - 2 * INNER_;
        const int h = cc >> 6, d = cc & 63;
        vtw[(((size_t)b * HEADS_ + h) * DH + d) * NSEQ + n] = (bf16_t)v;
      }
    }
  }
}

// ---------------- flash attention with relative position bias ----------------
// grid = BH * (NSEQ/128); 8 waves, each owns a 16-query strip.
__global__ __launch_bounds__(256) void attn_kernel(
    const bf16_t* __restrict__ qw, const bf16_t* __restrict__ kw,
    const bf16_t* __restrict__ vtw, const bf16_t* __restrict__ ptb,
    bf16_t* __restrict__ ao) {
  __shared__ bf16_t sbuf[8][16][160];  // per-wave staging: qp window, then P

  const int lane = threadIdx.x & 31;
  const int wave = threadIdx.x >> 5;
  const int ln = lane & 15;
  const int g = lane >> 4;

  const int bx = blockIdx.x;
  const int bh = bx >> 3;
  const int b = bh >> 3;
  const int h = bh & 7;
  const int i0 = (bx & 7) * 128 + wave * 16;

  const bf16_t* qbase = qw + (size_t)bh * NSEQ * DH;
  const bf16_t* kbase = kw + (size_t)bh * NSEQ * DH;
  const bf16_t* vbase = vtw + (size_t)bh * DH * NSEQ;

  const bf16x16 aq0 = load_a_frag(qbase, DH, i0, 0, lane);
  const bf16x16 aq1 = load_a_frag(qbase, DH, i0, 32, lane);

  f32x8 O[4];
#pragma unroll
  for (int t = 0; t < 4; ++t) O[t] = f32x8_zero();
  float mrun[8], lrun[8];
#pragma unroll
  for (int r = 0; r < 8; ++r) { mrun[r] = -1e30f; lrun[r] = 0.f; }

  for (int j0 = 0; j0 < NSEQ; j0 += 128) {
    // clip(i-j,-512,512)+512 over this tile spans <=144 table rows
    int w0 = (i0 - j0) + 385;
    w0 = w0 < 0 ? 0 : (w0 > (NPOS - 144) ? (NPOS - 144) : w0);

    // qp = Q @ pos_table[w0:w0+144]^T  -> stage to LDS (bf16)
#pragma unroll
    for (int t = 0; t < 9; ++t) {
      f32x8 c = f32x8_zero();
      bf16x16 b0 = load_b_frag(ptb, DH, w0 + 16 * t, 0, lane);
      bf16x16 b1 = load_b_frag(ptb, DH, w0 + 16 * t, 32, lane);
      c = wmma_bf16(aq0, b0, c);
      c = wmma_bf16(aq1, b1, c);
#pragma unroll
      for (int r = 0; r < 8; ++r) sbuf[wave][r + 8 * g][16 * t + ln] = (bf16_t)c[r];
    }

    // S = Q @ K^T (16 x 128)
    f32x8 S[8];
#pragma unroll
    for (int t = 0; t < 8; ++t) {
      f32x8 c = f32x8_zero();
      bf16x16 b0 = load_b_frag(kbase, DH, j0 + 16 * t, 0, lane);
      bf16x16 b1 = load_b_frag(kbase, DH, j0 + 16 * t, 32, lane);
      c = wmma_bf16(aq0, b0, c);
      c = wmma_bf16(aq1, b1, c);
      S[t] = c;
    }
    asm volatile("s_wait_dscnt 0" ::: "memory");

    // add relative-position bias, track row max
    float rm[8];
#pragma unroll
    for (int r = 0; r < 8; ++r) rm[r] = -1e30f;
#pragma unroll
    for (int t = 0; t < 8; ++t) {
#pragma unroll
      for (int r = 0; r < 8; ++r) {
        const int i = i0 + r + 8 * g;
        const int j = j0 + 16 * t + ln;
        int p = i - j + 512;
        p = p < 0 ? 0 : (p > 1024 ? 1024 : p);
        p -= w0;
        const float s = S[t][r] + (float)sbuf[wave][r + 8 * g][p];
        S[t][r] = s;
        rm[r] = fmaxf(rm[r], s);
      }
    }
#pragma unroll
    for (int r = 0; r < 8; ++r) {
      float v = rm[r];
      v = fmaxf(v, __shfl_xor(v, 1));
      v = fmaxf(v, __shfl_xor(v, 2));
      v = fmaxf(v, __shfl_xor(v, 4));
      v = fmaxf(v, __shfl_xor(v, 8));
      rm[r] = v;
    }

    float alpha[8], rs[8];
#pragma unroll
    for (int r = 0; r < 8; ++r) {
      const float mn = fmaxf(mrun[r], rm[r]);
      alpha[r] = __expf(mrun[r] - mn);
      mrun[r] = mn;
      rs[r] = 0.f;
    }
#pragma unroll
    for (int t = 0; t < 8; ++t) {
#pragma unroll
      for (int r = 0; r < 8; ++r) {
        const float p = __expf(S[t][r] - mrun[r]);
        S[t][r] = p;
        rs[r] += p;
      }
    }
#pragma unroll
    for (int r = 0; r < 8; ++r) {
      float v = rs[r];
      v += __shfl_xor(v, 1);
      v += __shfl_xor(v, 2);
      v += __shfl_xor(v, 4);
      v += __shfl_xor(v, 8);
      lrun[r] = lrun[r] * alpha[r] + v;
    }
#pragma unroll
    for (int t = 0; t < 4; ++t) {
#pragma unroll
      for (int r = 0; r < 8; ++r) O[t][r] *= alpha[r];
    }

    // stage P (C-layout -> LDS, bf16), then O += P @ V
#pragma unroll
    for (int t = 0; t < 8; ++t) {
#pragma unroll
      for (int r = 0; r < 8; ++r)
        sbuf[wave][r + 8 * g][16 * t + ln] = (bf16_t)S[t][r];
    }
    asm volatile("s_wait_dscnt 0" ::: "memory");

    bf16x16 AP[4];
#pragma unroll
    for (int kc = 0; kc < 4; ++kc)
      AP[kc] = load_a_frag(&sbuf[wave][0][0], 160, 0, 32 * kc, lane);
#pragma unroll
    for (int t2 = 0; t2 < 4; ++t2) {
#pragma unroll
      for (int kc = 0; kc < 4; ++kc) {
        bf16x16 bv = load_b_frag(vbase, NSEQ, 16 * t2, j0 + 32 * kc, lane);
        O[t2] = wmma_bf16(AP[kc], bv, O[t2]);
      }
    }
  }

  // normalize and store attention output as [b][n][h*64+d] (bf16)
#pragma unroll
  for (int t2 = 0; t2 < 4; ++t2) {
#pragma unroll
    for (int r = 0; r < 8; ++r) {
      const int i = i0 + r + 8 * g;
      const int d = 16 * t2 + ln;
      ao[((size_t)b * NSEQ + i) * INNER_ + h * DH + d] = (bf16_t)(O[t2][r] / lrun[r]);
    }
  }
}

// ---------------- output projection ----------------
__global__ __launch_bounds__(256) void out_gemm_kernel(
    const bf16_t* __restrict__ ao, const bf16_t* __restrict__ wot,
    const float* __restrict__ bo, float* __restrict__ out) {
  const int lane = threadIdx.x & 31;
  const int wave = threadIdx.x >> 5;
  const int ln = lane & 15;
  const int g = lane >> 4;
  const int m0 = blockIdx.x * 128 + wave * 16;

  bf16x16 A[16];
#pragma unroll
  for (int kc = 0; kc < 16; ++kc) A[kc] = load_a_frag(ao, INNER_, m0, 32 * kc, lane);

  for (int ct = 0; ct < 32; ++ct) {
    f32x8 c = f32x8_zero();
#pragma unroll
    for (int kc = 0; kc < 16; ++kc) {
      bf16x16 bf = load_b_frag(wot, INNER_, ct * 16, 32 * kc, lane);
      c = wmma_bf16(A[kc], bf, c);
    }
    const int col = ct * 16 + ln;
    const float bias = bo[col];
#pragma unroll
    for (int r = 0; r < 8; ++r)
      out[(size_t)(m0 + r + 8 * g) * DIMM + col] = c[r] + bias;
  }
}

extern "C" void kernel_launch(void* const* d_in, const int* in_sizes, int n_in,
                              void* d_out, int out_size, void* d_ws, size_t ws_size,
                              hipStream_t stream) {
  const float* x = (const float*)d_in[0];
  const float* Wq = (const float*)d_in[1];
  const float* Wkv = (const float*)d_in[2];
  const float* Wo = (const float*)d_in[3];
  const float* bo = (const float*)d_in[4];
  const float* pt = (const float*)d_in[5];
  float* out = (float*)d_out;

  char* ws = (char*)d_ws;
  size_t off = 0;
  auto alloc = [&](size_t bytes) -> void* {
    void* p = ws + off;
    off = (off + bytes + 255) & ~(size_t)255;
    return p;
  };
  bf16_t* xb = (bf16_t*)alloc((size_t)8192 * 512 * 2);     // x bf16
  bf16_t* wt = (bf16_t*)alloc((size_t)1536 * 512 * 2);     // [Wq|Wkv]^T bf16
  bf16_t* wot = (bf16_t*)alloc((size_t)512 * 512 * 2);     // Wo^T bf16
  bf16_t* ptb = (bf16_t*)alloc((size_t)NPOS * DH * 2);     // pos_table bf16
  bf16_t* qw = (bf16_t*)alloc((size_t)64 * NSEQ * DH * 2); // q (pre-scaled)
  bf16_t* kw = (bf16_t*)alloc((size_t)64 * NSEQ * DH * 2); // k
  bf16_t* vtw = (bf16_t*)alloc((size_t)64 * DH * NSEQ * 2);// v transposed
  bf16_t* ao = (bf16_t*)alloc((size_t)8192 * 512 * 2);     // attention out

  const int nx = 8192 * 512;
  const int npt = NPOS * DH;
  cvt_f32_bf16_kernel<<<(nx + 255) / 256, 256, 0, stream>>>(x, xb, nx);
  cvt_wqkv_kernel<<<(1536 * 512 + 255) / 256, 256, 0, stream>>>(Wq, Wkv, wt);
  cvt_wo_kernel<<<(512 * 512 + 255) / 256, 256, 0, stream>>>(Wo, wot);
  cvt_f32_bf16_kernel<<<(npt + 255) / 256, 256, 0, stream>>>(pt, ptb, npt);

  qkv_gemm_kernel<<<64, 256, 0, stream>>>(xb, wt, qw, kw, vtw);
  attn_kernel<<<64 * (NSEQ / 128), 256, 0, stream>>>(qw, kw, vtw, ptb, ao);
  out_gemm_kernel<<<64, 256, 0, stream>>>(ao, wot, bo, out);
}
